// VectorQuantizer_59407987638586
// MI455X (gfx1250) — matrix operand order repaired and checked by reading, not compiled, not run on hardware
//
#include <hip/hip_runtime.h>
#include <stdint.h>

// ---------------- problem constants ----------------
#define N_ROWS   131072            // B*H*W = 32*64*64
#define DIM      64                // embedding dim (== C)
#define KCODES   1024              // codebook entries
#define HW_MASK  4095              // H*W - 1
#define QOUT_ELEMS (N_ROWS * DIM)  // 8388608 floats (quantized output)

#define FRAG_BYTES  131072                     // 64 chunks * 2 wmma * 32 lanes * 32B
#define WNORM_OFF   FRAG_BYTES
#define STAGE_BYTES (FRAG_BYTES + KCODES * 4)  // 135168 (8B-multiple)
#define IDX_SM_OFF  STAGE_BYTES
#define SMEM_BYTES  (STAGE_BYTES + 8 * 16 * 4) // + per-wave argmin rows

typedef __bf16 bf16_t;
typedef __attribute__((ext_vector_type(16))) __bf16 v16bf;
typedef __attribute__((ext_vector_type(8)))  float  v8f;
typedef __attribute__((ext_vector_type(2)))  float  f32x2;
typedef __attribute__((ext_vector_type(4)))  unsigned int u32x4;
typedef __attribute__((ext_vector_type(8)))  int    i32x8;
typedef __attribute__((ext_vector_type(4)))  int    i32x4;

// ---------------------------------------------------------------------------
// Prep: swizzle f32 codebook into per-lane bf16 WMMA B-fragments + ||w||^2.
// B-fragment layout used here for v_wmma_f32_16x16x32_bf16 (contiguous k, same as A):
//   lane = 16*hi + l : column N = l ; VGPR v half j holds K = 16*hi + 2v + j
// Fragment (kc, w) covers global d in [32w, 32w+32) for codes [16kc, 16kc+16).
// Fragments are stored with 32B lane stride (bank-conflict-free for b128).
// ---------------------------------------------------------------------------
__global__ __launch_bounds__(256)
void vq_prep(const float* __restrict__ w, bf16_t* __restrict__ frag,
             float* __restrict__ wnorm) {
  int gid = blockIdx.x * blockDim.x + threadIdx.x;
  if (gid < 64 * 2 * 32) {               // fragment swizzle lanes
    int lane = gid & 31, group = gid >> 5;
    int kc = group >> 1, wi = group & 1;
    int l = lane & 15, hi = lane >> 4;
    const float* row = w + (kc * 16 + l) * DIM + wi * 32;
    bf16_t* dst = frag + gid * 16;
#pragma unroll
    for (int v = 0; v < 8; ++v)
#pragma unroll
      for (int j = 0; j < 2; ++j) {
        int kl = 16 * hi + 2 * v + j;    // K index within the 32-wide slab
        dst[2 * v + j] = (bf16_t)row[kl];
      }
  } else if (gid < 64 * 2 * 32 + KCODES) {
    int code = gid - 64 * 2 * 32;
    const float* row = w + code * DIM;
    float s = 0.f;
#pragma unroll
    for (int i = 0; i < DIM; ++i) s += row[i] * row[i];
    wnorm[code] = s;
  }
}

// ---------------------------------------------------------------------------
// Main: fused GEMM + argmin + gather.  1 block = 8 waves = 128 rows.
// Pipeline: issue TDM stage (async) -> load A fragments from HBM (overlaps
// the TDM transfer) -> wait tensorcnt + barrier -> WMMA sweep -> gather.
// ---------------------------------------------------------------------------
__global__ __launch_bounds__(256)
void vq_main(const float* __restrict__ z, const float* __restrict__ w,
             const unsigned char* __restrict__ ws, float* __restrict__ out) {
  extern __shared__ unsigned char smem[];
  const int tid  = threadIdx.x;
  const int wave = tid >> 5;
  const int lane = tid & 31;
  const int l    = lane & 15;
  const int hi   = lane >> 4;
  const int rowBase = blockIdx.x * 128 + wave * 16;

  // ---- kick off async staging of pre-swizzled codebook into LDS ----
#if __has_builtin(__builtin_amdgcn_tensor_load_to_lds)
  if (wave == 0) {
    unsigned long long ga = (unsigned long long)(uintptr_t)ws;
    unsigned int lds = (unsigned int)(uintptr_t)smem;
    const unsigned elems = STAGE_BYTES >> 3;           // 8-byte elements
    u32x4 g0;
    g0[0] = 1u;                                        // count=1, load, user D#
    g0[1] = lds;                                       // lds_addr
    g0[2] = (unsigned)(ga & 0xFFFFFFFFu);              // global_addr[31:0]
    g0[3] = (unsigned)((ga >> 32) & 0x1FFFFFFu) | (2u << 30); // addr hi | type=2
    i32x8 g1;
    g1[0] = (3 << 16);                                 // data_size = 8B
    g1[1] = (int)((elems & 0xFFFFu) << 16);            // tensor_dim0 lo
    g1[2] = (int)((elems >> 16) & 0xFFFFu) | (1 << 16);// tensor_dim0 hi | dim1=1
    g1[3] = (int)((elems & 0xFFFFu) << 16);            // tile_dim0 = whole image
    g1[4] = 0;                                         // tile_dim1/2 unused
    g1[5] = (int)elems;                                // dim0_stride lo
    g1[6] = 0;
    g1[7] = 0;
    i32x4 zero4 = {0, 0, 0, 0};
#if defined(__clang_major__) && (__clang_major__ >= 23)
    __builtin_amdgcn_tensor_load_to_lds(g0, g1, zero4, zero4,
                                        (i32x8){0, 0, 0, 0, 0, 0, 0, 0}, 0);
#else
    __builtin_amdgcn_tensor_load_to_lds(g0, g1, zero4, zero4, 0);
#endif
  }
#endif

  // ---- load A fragments while the TDM runs: 16 rows x 64 ch, f32 -> bf16 ----
  // A layout used here (contiguous k, same as B): lane l/hi holds row M=l;
  //   VGPR v half j holds K = 16*hi + 2v + j
  const int n0 = rowBase + l;
  const float* zb = z + (((size_t)(n0 >> 12)) << 18) + (n0 & HW_MASK);
  v16bf a0, a1;
#pragma unroll
  for (int v = 0; v < 8; ++v)
#pragma unroll
    for (int j = 0; j < 2; ++j) {
      int kk = 16 * hi + 2 * v + j;
      a0[2 * v + j] = (bf16_t)__builtin_nontemporal_load(zb + ((size_t)kk << 12));
      a1[2 * v + j] = (bf16_t)__builtin_nontemporal_load(zb + ((size_t)(kk + 32) << 12));
    }

  // ---- staging complete before anyone touches LDS ----
#if __has_builtin(__builtin_amdgcn_tensor_load_to_lds)
  if (wave == 0) __builtin_amdgcn_s_wait_tensorcnt(0);
#else
  for (int i = tid; i < (STAGE_BYTES >> 3); i += 256)
    ((unsigned long long*)smem)[i] = ((const unsigned long long*)ws)[i];
#endif
  __syncthreads();

  const v16bf* fragSm = (const v16bf*)smem;            // 32B per lane per frag
  const float* wnSm   = (const float*)(smem + WNORM_OFF);
  int* idxSm          = (int*)(smem + IDX_SM_OFF) + wave * 16;

  // ---- sweep 1024 codes in 16-wide chunks: 2 WMMAs + argmin update ----
  float best[8];
  int   bidx[8];
#pragma unroll
  for (int v = 0; v < 8; ++v) { best[v] = -3.402823466e38f; bidx[v] = 0; }
  const v8f zeroacc = {0, 0, 0, 0, 0, 0, 0, 0};

#pragma unroll 2
  for (int kc = 0; kc < 64; ++kc) {
    const float h = -0.5f * wnSm[kc * 16 + l];         // score = dot - 0.5||w||^2
    const v16bf b0 = fragSm[(kc * 2 + 0) * 32 + lane];
    const v16bf b1 = fragSm[(kc * 2 + 1) * 32 + lane];
    v8f acc = __builtin_amdgcn_wmma_f32_16x16x32_bf16(
        false, a0, false, b0, (short)0, zeroacc, false, false);
    acc = __builtin_amdgcn_wmma_f32_16x16x32_bf16(
        false, a1, false, b1, (short)0, acc, false, false);
    const int code = kc * 16 + l;
#pragma unroll
    for (int v = 0; v < 8; ++v) {
      float s = acc[v] + h;
      bool g = s > best[v];                            // strict > keeps 1st index
      best[v] = g ? s : best[v];
      bidx[v] = g ? code : bidx[v];
    }
  }

  // ---- reduce across the 16 columns (stays within each hi half) ----
#pragma unroll
  for (int off = 8; off > 0; off >>= 1) {
#pragma unroll
    for (int v = 0; v < 8; ++v) {
      float ob = __shfl_xor(best[v], off, 32);
      int   oi = __shfl_xor(bidx[v], off, 32);
      if (ob > best[v] || (ob == best[v] && oi < bidx[v])) {
        best[v] = ob; bidx[v] = oi;
      }
    }
  }
  if (l == 0) {                                        // lanes 0 & 16: rows hi*8+v
#pragma unroll
    for (int v = 0; v < 8; ++v) {
      idxSm[hi * 8 + v] = bidx[v];
      out[QOUT_ELEMS + rowBase + hi * 8 + v] = (float)bidx[v];
    }
  }
  __syncthreads();

  // ---- gather winning f32 codebook rows (exact payload) ----
#pragma unroll 4
  for (int r = 0; r < 16; ++r) {
    const int idx = idxSm[r];
    const f32x2* src = (const f32x2*)(w + (size_t)idx * DIM);
    f32x2* dst = (f32x2*)(out + (size_t)(rowBase + r) * DIM);
    __builtin_nontemporal_store(src[lane], dst + lane);
  }
}

// ---------------------------------------------------------------------------
extern "C" void kernel_launch(void* const* d_in, const int* in_sizes, int n_in,
                              void* d_out, int out_size, void* d_ws, size_t ws_size,
                              hipStream_t stream) {
  (void)in_sizes; (void)n_in; (void)out_size; (void)ws_size;
  const float* z      = (const float*)d_in[0];
  const float* weight = (const float*)d_in[1];
  float* out = (float*)d_out;
  bf16_t* frag = (bf16_t*)d_ws;
  float* wnorm = (float*)((unsigned char*)d_ws + WNORM_OFF);

  vq_prep<<<dim3(20), dim3(256), 0, stream>>>(weight, frag, wnorm);
  vq_main<<<dim3(N_ROWS / 128), dim3(256), SMEM_BYTES, stream>>>(
      z, weight, (const unsigned char*)d_ws, out);
}
